// Nystromformer_28149215658657
// MI455X (gfx1250) — matrix operand order repaired and checked by reading, not compiled, not run on hardware
//
#include <hip/hip_runtime.h>
#include <hip/hip_bf16.h>

typedef __attribute__((ext_vector_type(16))) __bf16 v16bf;
typedef __attribute__((ext_vector_type(8)))  float  v8f;
typedef __attribute__((ext_vector_type(2)))  float  v2f;

#define BH    48
#define SEQ   4096
#define DIM   64
#define M_LM  64
#define SCALE 0.125f            // 1/sqrt(64), exact power of two
#define LOG2E 1.44269504088896f
#define NCHUNK 8
#define PART_STRIDE (M_LM*DIM + 2*M_LM)   // 4224 floats: pbv[64][64], m[64], sum[64]

// ---- workspace layout (float offsets) ----
#define WS_QLM   0
#define WS_KLM   (BH*M_LM*DIM)
#define WS_S     (2*BH*M_LM*DIM)
#define WS_BV    (3*BH*M_LM*DIM)
#define WS_ABVT  (4*BH*M_LM*DIM)
#define WS_PART  (5*BH*M_LM*DIM)
#define WS_COEF  (WS_PART + BH*NCHUNK*PART_STRIDE)

// ================= WMMA wrappers =================
__device__ inline v8f z8() {
  v8f z;
  #pragma unroll
  for (int i = 0; i < 8; ++i) z[i] = 0.0f;
  return z;
}

__device__ inline v8f wmma_bf(v16bf a, v16bf b, v8f c) {
  return __builtin_amdgcn_wmma_f32_16x16x32_bf16(false, a, false, b, (short)0, c, false, false);
}
__device__ inline v8f wmma_f32(v2f a, v2f b, v8f c) {
  return __builtin_amdgcn_wmma_f32_16x16x4_f32(false, a, false, b, (short)0, c, false, false);
}

// A-operand (16x32 bf16): lane holds row M=lane&15; element e -> K=(e&7)+(e>>3)*16+half*8 (+kk)
__device__ inline v16bf ldA(const __bf16* row, int kk, int half) {
  v16bf a; const __bf16* p = row + kk + half * 8;
  #pragma unroll
  for (int e = 0; e < 8; ++e) { a[e] = p[e]; a[e + 8] = p[16 + e]; }
  return a;
}
// B-operand (32x16 bf16): lane holds col N=lane&15; element e -> K=(e&7)+(e>>3)*16+half*8 (+kk)
__device__ inline v16bf ldB(const __bf16* row, int kk, int half) {
  v16bf b; const __bf16* p = row + kk + half * 8;
  #pragma unroll
  for (int e = 0; e < 8; ++e) { b[e] = p[e]; b[e + 8] = p[16 + e]; }
  return b;
}

// ---- DPP16 butterfly reductions confined to 16-lane rows (VALU only, no LDS) ----
// quad_perm[1,0,3,2]=0xB1 (xor1), quad_perm[2,3,0,1]=0x4E (xor2),
// row_half_mirror=0x141 (lane^7 == effective xor4 once quads uniform),
// row_mirror=0x140 (lane^15 == effective xor8 once 8-groups uniform).
#define DPP_F(v, ctrl) \
  __int_as_float(__builtin_amdgcn_mov_dpp(__float_as_int(v), (ctrl), 0xf, 0xf, true))

__device__ inline float redmax16(float v) {
  v = fmaxf(v, DPP_F(v, 0xB1));
  v = fmaxf(v, DPP_F(v, 0x4E));
  v = fmaxf(v, DPP_F(v, 0x141));
  v = fmaxf(v, DPP_F(v, 0x140));
  return v;
}
__device__ inline float redsum16(float v) {
  v += DPP_F(v, 0xB1);
  v += DPP_F(v, 0x4E);
  v += DPP_F(v, 0x141);
  v += DPP_F(v, 0x140);
  return v;
}

// 64x64x64 f32 matmul in LDS via v_wmma_f32_16x16x4_f32; 8 waves x 2 tiles.
// If BT: B is stored transposed (B[k][n] = Bm[n*64+k]).
__device__ inline void mm64(float* C, const float* A, const float* Bm, bool BT, int tid) {
  int wave = tid >> 5, lane = tid & 31, half = lane >> 4, ln = lane & 15;
  #pragma unroll
  for (int tt = 0; tt < 2; ++tt) {
    int t = wave * 2 + tt, mi = t >> 2, ni = t & 3;
    v8f acc = z8();
    const float* arow = A + (mi * 16 + ln) * 64;
    #pragma unroll
    for (int k0 = 0; k0 < 64; k0 += 4) {
      v2f a, b;
      a[0] = arow[k0 + half * 2]; a[1] = arow[k0 + half * 2 + 1];
      if (BT) {
        const float* brow = Bm + (ni * 16 + ln) * 64;
        b[0] = brow[k0 + half * 2]; b[1] = brow[k0 + half * 2 + 1];
      } else {
        b[0] = Bm[(k0 + half * 2) * 64 + ni * 16 + ln];
        b[1] = Bm[(k0 + half * 2 + 1) * 64 + ni * 16 + ln];
      }
      acc = wmma_f32(a, b, acc);
    }
    #pragma unroll
    for (int r = 0; r < 8; ++r)
      C[(mi * 16 + r + half * 8) * 64 + ni * 16 + ln] = acc[r];
  }
}

// ================= K1: landmarks (segment means) =================
__global__ __launch_bounds__(128) void nys_k1_landmarks(
    const float* __restrict__ Q, const float* __restrict__ Kg,
    float* __restrict__ qlm, float* __restrict__ klm, int* __restrict__ coefbits) {
  int seg = blockIdx.x, bh = blockIdx.y, tid = threadIdx.x;
  const float* src = (tid < 64) ? Q : Kg;
  float* dst = (tid < 64) ? qlm : klm;
  int col = tid & 63;
  size_t base = (size_t)bh * SEQ * DIM + (size_t)seg * 64 * DIM;
  float s = 0.f;
  #pragma unroll 4
  for (int r = 0; r < 64; ++r) s += src[base + (size_t)r * DIM + col];
  dst[bh * M_LM * DIM + seg * DIM + col] = s * (1.0f / 64.0f);
  if (seg == 0 && bh == 0 && tid == 0) { coefbits[0] = 0; coefbits[1] = 0; }
}

// ======== K2a: partial bv = softmax(q_lm K^T) V over a 512-wide s-chunk ========
__global__ __launch_bounds__(128) void nys_k2a_bv_partial(
    const float* __restrict__ Kg, const float* __restrict__ Vg,
    const float* __restrict__ qlm, float* __restrict__ part) {
  __shared__ __bf16 qlm_bf[M_LM * DIM];  // 8 KB  [j][d], pre-scaled by 0.125
  __shared__ __bf16 Kbf[32 * DIM];       // 4 KB  [s'][d]
  __shared__ __bf16 VT[DIM * 32];        // 4 KB  [d][s'] (transposed for B-layout)
  __shared__ __bf16 Pld[4][16 * 32];     // 4 KB  per-wave P tile [m'][s']
  int chunk = blockIdx.x, bh = blockIdx.y, tid = threadIdx.x;
  int wave = tid >> 5, lane = tid & 31, half = lane >> 4, ln = lane & 15;

  // pre-scale by SCALE (power of two -> exact in bf16); logits come out scaled
  for (int i = tid; i < M_LM * DIM; i += 128)
    qlm_bf[i] = (__bf16)(qlm[bh * M_LM * DIM + i] * SCALE);

  v8f pbv[4]; float rmax[8], rsum[8];
  #pragma unroll
  for (int t = 0; t < 4; ++t) pbv[t] = z8();
  #pragma unroll
  for (int r = 0; r < 8; ++r) { rmax[r] = -1e30f; rsum[r] = 0.f; }

  size_t hb = (size_t)bh * SEQ * DIM;
  for (int it = 0; it < 16; ++it) {
    int s0 = chunk * 512 + it * 32;
    __syncthreads();  // protect Kbf/VT from previous iteration readers
    for (int i = tid; i < 32 * DIM; i += 128)
      Kbf[i] = (__bf16)Kg[hb + (size_t)s0 * DIM + i];
    for (int i = tid; i < 32 * DIM; i += 128) {
      int sp = i >> 6, d = i & 63;
      VT[d * 32 + sp] = (__bf16)Vg[hb + (size_t)(s0 + sp) * DIM + d];
    }
    __syncthreads();
    // logits (already scaled) 16x32 for this wave's landmark tile
    v8f accL0 = z8(), accL1 = z8();
    const __bf16* arow = qlm_bf + (wave * 16 + ln) * DIM;
    #pragma unroll
    for (int kk = 0; kk < 64; kk += 32) {
      v16bf a = ldA(arow, kk, half);
      accL0 = wmma_bf(a, ldB(Kbf + ln * DIM, kk, half), accL0);
      accL1 = wmma_bf(a, ldB(Kbf + (16 + ln) * DIM, kk, half), accL1);
    }
    // online softmax update + build P (bf16)
    #pragma unroll
    for (int r = 0; r < 8; ++r) {
      float l0 = accL0[r], l1 = accL1[r];
      float tm = redmax16(fmaxf(l0, l1));
      float mnew = fmaxf(rmax[r], tm);
      float corr = exp2f((rmax[r] - mnew) * LOG2E);
      rmax[r] = mnew;
      float p0 = exp2f((l0 - mnew) * LOG2E), p1 = exp2f((l1 - mnew) * LOG2E);
      rsum[r] = rsum[r] * corr + redsum16(p0 + p1);
      #pragma unroll
      for (int t = 0; t < 4; ++t) pbv[t][r] *= corr;
      Pld[wave][(r + half * 8) * 32 + ln] = (__bf16)p0;
      Pld[wave][(r + half * 8) * 32 + 16 + ln] = (__bf16)p1;
    }
    __syncthreads();
    // pbv += P(16x32) @ V(32x64)
    v16bf pa = ldA(&Pld[wave][ln * 32], 0, half);
    #pragma unroll
    for (int t = 0; t < 4; ++t)
      pbv[t] = wmma_bf(pa, ldB(VT + (t * 16 + ln) * 32, 0, half), pbv[t]);
  }
  // write partials
  float* pb = part + (size_t)(bh * NCHUNK + chunk) * PART_STRIDE;
  #pragma unroll
  for (int t = 0; t < 4; ++t) {
    #pragma unroll
    for (int r = 0; r < 8; ++r)
      pb[(wave * 16 + r + half * 8) * DIM + t * 16 + ln] = pbv[t][r];
  }
  if (ln == 0) {
    #pragma unroll
    for (int r = 0; r < 8; ++r) {
      int row = wave * 16 + r + half * 8;
      pb[M_LM * DIM + row] = rmax[r];
      pb[M_LM * DIM + M_LM + row] = rsum[r];
    }
  }
}

// ================= K2b: combine chunk partials -> bv =================
__global__ __launch_bounds__(256) void nys_k2b_combine(
    const float* __restrict__ part, float* __restrict__ bv) {
  __shared__ float mstar[M_LM], sstar[M_LM];
  int bh = blockIdx.x, tid = threadIdx.x;
  const float* pb0 = part + (size_t)bh * NCHUNK * PART_STRIDE;
  if (tid < M_LM) {
    int j = tid;
    float m = -1e30f;
    for (int c = 0; c < NCHUNK; ++c) m = fmaxf(m, pb0[c * PART_STRIDE + M_LM * DIM + j]);
    float s = 0.f;
    for (int c = 0; c < NCHUNK; ++c) {
      float mc = pb0[c * PART_STRIDE + M_LM * DIM + j];
      s += pb0[c * PART_STRIDE + M_LM * DIM + M_LM + j] * exp2f((mc - m) * LOG2E);
    }
    mstar[j] = m; sstar[j] = s;
  }
  __syncthreads();
  int j = tid >> 2, d0 = (tid & 3) * 16;
  float inv = 1.0f / sstar[j];
  for (int dd = 0; dd < 16; ++dd) {
    int d = d0 + dd; float acc = 0.f;
    for (int c = 0; c < NCHUNK; ++c) {
      float w = exp2f((pb0[c * PART_STRIDE + M_LM * DIM + j] - mstar[j]) * LOG2E);
      acc += pb0[c * PART_STRIDE + j * DIM + d] * w;
    }
    bv[bh * M_LM * DIM + j * DIM + d] = acc * inv;
  }
}

// ====== K3a: S = softmax(q_lm k_lm^T * scale); global init-coef atomics ======
__global__ __launch_bounds__(256) void nys_k3a_soft(
    const float* __restrict__ qlm, const float* __restrict__ klm,
    float* __restrict__ Sws, int* __restrict__ coefbits) {
  __shared__ float Al[4096], Bl[4096], Sl[4096];  // 48 KB
  int bh = blockIdx.x, tid = threadIdx.x;
  for (int i = tid; i < 4096; i += 256) { Al[i] = qlm[bh * 4096 + i]; Bl[i] = klm[bh * 4096 + i]; }
  __syncthreads();
  mm64(Sl, Al, Bl, true, tid);  // logits = q_lm @ k_lm^T (f32 WMMA)
  __syncthreads();
  if (tid < 64) {
    float* row = Sl + tid * 64;
    float m = -1e30f;
    for (int c = 0; c < 64; ++c) m = fmaxf(m, row[c] * SCALE);
    float s = 0.f;
    for (int c = 0; c < 64; ++c) { float e = exp2f((row[c] * SCALE - m) * LOG2E); row[c] = e; s += e; }
    float inv = 1.0f / s;
    for (int c = 0; c < 64; ++c) row[c] *= inv;
  }
  __syncthreads();
  for (int i = tid; i < 4096; i += 256) Sws[bh * 4096 + i] = Sl[i];
  if (tid < 64) {
    float cs = 0.f, rs = 0.f;
    for (int r = 0; r < 64; ++r) cs += Sl[r * 64 + tid];
    for (int c = 0; c < 64; ++c) rs += Sl[tid * 64 + c];
    Al[tid] = cs; Al[64 + tid] = rs;
  }
  __syncthreads();
  if (tid == 0) {
    float mc = 0.f, mr = 0.f;
    for (int j = 0; j < 64; ++j) { mc = fmaxf(mc, Al[j]); mr = fmaxf(mr, Al[64 + j]); }
    atomicMax(&coefbits[0], __float_as_int(mc));   // valid: values >= 0
    atomicMax(&coefbits[1], __float_as_int(mr));
  }
}

// ====== K3b: iterative pinv (6 Newton steps) + abv = A_tilde @ bv ======
__global__ __launch_bounds__(256) void nys_k3b_pinv(
    const float* __restrict__ Sws, const float* __restrict__ bvw,
    float* __restrict__ abvT, const int* __restrict__ coefbits) {
  extern __shared__ float sm[];
  float* S = sm; float* out = sm + 4096; float* mo = sm + 8192;
  float* w1 = sm + 12288; float* w2 = sm + 16384;   // 5 x 16KB = 80KB dyn LDS
  int bh = blockIdx.x, tid = threadIdx.x;
  float icoef = 1.0f / (__int_as_float(coefbits[0]) * __int_as_float(coefbits[1]));
  for (int i = tid; i < 4096; i += 256) S[i] = Sws[bh * 4096 + i];
  __syncthreads();
  for (int i = tid; i < 4096; i += 256) { int r = i >> 6, c = i & 63; out[i] = S[c * 64 + r] * icoef; }
  __syncthreads();
  for (int iter = 0; iter < 6; ++iter) {
    mm64(mo, S, out, false, tid); __syncthreads();
    for (int i = tid; i < 4096; i += 256) { int r = i >> 6, c = i & 63; w1[i] = (r == c ? 7.f : 0.f) - mo[i]; }
    __syncthreads();
    mm64(w2, mo, w1, false, tid); __syncthreads();
    for (int i = tid; i < 4096; i += 256) { int r = i >> 6, c = i & 63; w2[i] = (r == c ? 15.f : 0.f) - w2[i]; }
    __syncthreads();
    mm64(w1, mo, w2, false, tid); __syncthreads();
    for (int i = tid; i < 4096; i += 256) { int r = i >> 6, c = i & 63; w1[i] = (r == c ? 13.f : 0.f) - w1[i]; }
    __syncthreads();
    mm64(mo, out, w1, false, tid); __syncthreads();
    for (int i = tid; i < 4096; i += 256) out[i] = 0.25f * mo[i];
    __syncthreads();
  }
  for (int i = tid; i < 4096; i += 256) w1[i] = bvw[bh * 4096 + i];
  __syncthreads();
  mm64(w2, out, w1, false, tid); __syncthreads();
  // store transposed: abvT[d][j] = abv[j][d]
  for (int i = tid; i < 4096; i += 256) {
    int j = i >> 6, d = i & 63;
    abvT[bh * 4096 + d * 64 + j] = w2[i];
  }
}

// ====== K4: out = softmax(Q k_lm^T) @ abv  (dominant-traffic kernel) ======
__global__ __launch_bounds__(256) void nys_k4_out(
    const float* __restrict__ Qg, const float* __restrict__ klm,
    const float* __restrict__ abvTw, float* __restrict__ Og) {
  __shared__ __bf16 klm_bf[M_LM * DIM];  // 8 KB  [j][d]
  __shared__ __bf16 abv_bf[DIM * M_LM];  // 8 KB  [d][j] (pre-transposed)
  __shared__ __bf16 Qbf[8][16 * DIM];    // 16 KB per-wave Q slab, pre-scaled by 0.125
  __shared__ __bf16 Fbf[8][16 * DIM];    // 16 KB per-wave F slab
  int chunk = blockIdx.x, bh = blockIdx.y, tid = threadIdx.x;
  int wave = tid >> 5, lane = tid & 31, half = lane >> 4, ln = lane & 15;

  for (int i = tid; i < 4096; i += 256) {
    klm_bf[i] = (__bf16)klm[bh * 4096 + i];
    abv_bf[i] = (__bf16)abvTw[bh * 4096 + i];
  }
  size_t qbase = (size_t)bh * SEQ * DIM + (size_t)(chunk * 128 + wave * 16) * DIM;
  const float4* q4 = (const float4*)(Qg + qbase);
  #pragma unroll
  for (int i = 0; i < 8; ++i) {
    float4 f = q4[i * 32 + lane];
    int o = (i * 32 + lane) * 4;
    Qbf[wave][o] = (__bf16)(f.x * SCALE); Qbf[wave][o + 1] = (__bf16)(f.y * SCALE);
    Qbf[wave][o + 2] = (__bf16)(f.z * SCALE); Qbf[wave][o + 3] = (__bf16)(f.w * SCALE);
  }
  __syncthreads();
  // GEMM1: logits (pre-scaled) = Qslab(16x64) @ k_lm^T(64x64)
  v8f acc[4];
  #pragma unroll
  for (int t = 0; t < 4; ++t) acc[t] = z8();
  const __bf16* arow = Qbf[wave] + ln * DIM;
  #pragma unroll
  for (int kk = 0; kk < 64; kk += 32) {
    v16bf a = ldA(arow, kk, half);
    #pragma unroll
    for (int t = 0; t < 4; ++t)
      acc[t] = wmma_bf(a, ldB(klm_bf + (t * 16 + ln) * DIM, kk, half), acc[t]);
  }
  // row softmax over 64 landmarks (DPP16 butterflies), write F bf16
  #pragma unroll
  for (int r = 0; r < 8; ++r) {
    float l0 = acc[0][r], l1 = acc[1][r], l2 = acc[2][r], l3 = acc[3][r];
    float m = redmax16(fmaxf(fmaxf(l0, l1), fmaxf(l2, l3)));
    float p0 = exp2f((l0 - m) * LOG2E), p1 = exp2f((l1 - m) * LOG2E);
    float p2 = exp2f((l2 - m) * LOG2E), p3 = exp2f((l3 - m) * LOG2E);
    float inv = 1.0f / redsum16(p0 + p1 + p2 + p3);
    int ro = (r + half * 8) * DIM + ln;
    Fbf[wave][ro] = (__bf16)(p0 * inv);
    Fbf[wave][ro + 16] = (__bf16)(p1 * inv);
    Fbf[wave][ro + 32] = (__bf16)(p2 * inv);
    Fbf[wave][ro + 48] = (__bf16)(p3 * inv);
  }
  __syncthreads();
  // GEMM2: out = F(16x64) @ abv(64x64)
  v8f oa[4];
  #pragma unroll
  for (int t = 0; t < 4; ++t) oa[t] = z8();
  const __bf16* frow = Fbf[wave] + ln * DIM;
  #pragma unroll
  for (int kk = 0; kk < 64; kk += 32) {
    v16bf a = ldA(frow, kk, half);
    #pragma unroll
    for (int t = 0; t < 4; ++t)
      oa[t] = wmma_bf(a, ldB(abv_bf + (t * 16 + ln) * DIM, kk, half), oa[t]);
  }
  float* orow = Og + qbase;
  #pragma unroll
  for (int t = 0; t < 4; ++t) {
    #pragma unroll
    for (int r = 0; r < 8; ++r)
      orow[(r + half * 8) * DIM + t * 16 + ln] = oa[t][r];
  }
}

// ================= launch =================
extern "C" void kernel_launch(void* const* d_in, const int* in_sizes, int n_in,
                              void* d_out, int out_size, void* d_ws, size_t ws_size,
                              hipStream_t stream) {
  (void)in_sizes; (void)n_in; (void)out_size; (void)ws_size;
  const float* Q = (const float*)d_in[0];
  const float* K = (const float*)d_in[1];
  const float* V = (const float*)d_in[2];
  float* ws = (float*)d_ws;
  float* out = (float*)d_out;
  float* qlm = ws + WS_QLM;
  float* klm = ws + WS_KLM;
  float* Sws = ws + WS_S;
  float* bv = ws + WS_BV;
  float* abvT = ws + WS_ABVT;
  float* part = ws + WS_PART;
  int* coefbits = (int*)(ws + WS_COEF);

  nys_k1_landmarks<<<dim3(64, BH), 128, 0, stream>>>(Q, K, qlm, klm, coefbits);
  nys_k2a_bv_partial<<<dim3(NCHUNK, BH), 128, 0, stream>>>(K, V, qlm, part);
  nys_k2b_combine<<<BH, 256, 0, stream>>>(part, bv);
  nys_k3a_soft<<<BH, 256, 0, stream>>>(qlm, klm, Sws, coefbits);
  nys_k3b_pinv<<<BH, 256, 5 * 4096 * sizeof(float), stream>>>(Sws, bv, abvT, coefbits);
  nys_k4_out<<<dim3(SEQ / 128, BH), 256, 0, stream>>>(Q, klm, abvT, out);
}
